// FastAutoencoder_89670327206072
// MI455X (gfx1250) — compile-verified, matrix operand order checked by reading, not run on hardware
//
#include <hip/hip_runtime.h>
#include <hip/hip_bf16.h>

// ---------------- Problem constants ----------------
#define BATCH   1024
#define DMODEL  2048
#define NDIRS   32768
#define TOPK    32
#define AUXK    256
#define DEADTH  256
#define EPSV    0.001f

// ---------------- CDNA5 WMMA / vector types ----------------
typedef __attribute__((ext_vector_type(16))) _Float16     v16h;
typedef __attribute__((ext_vector_type(8)))  float        v8f;
typedef __attribute__((ext_vector_type(4)))  unsigned int u32x4;
typedef __attribute__((ext_vector_type(4)))  float        f32x4;
typedef __attribute__((ext_vector_type(4)))  unsigned int v4u;
typedef __attribute__((ext_vector_type(8)))  int          v8i;
typedef __attribute__((ext_vector_type(4)))  int          v4i;

#define HAVE_TDM __has_builtin(__builtin_amdgcn_tensor_load_to_lds)
#define HAVE_WTC __has_builtin(__builtin_amdgcn_s_wait_tensorcnt)

union Frag {
    v16h     v;
    _Float16 h[16];
    u32x4    q[2];
};

// 16-bit A/B fragment layout (16x32): lane<16 -> K chunks {kb+0..7, kb+16..23},
// lane>=16 -> {kb+8..15, kb+24..31}; "base" already includes the lane-half +8.
__device__ __forceinline__ void load_frag_f16(Frag& f, const _Float16* base) {
    f.q[0] = *(const u32x4*)(const void*)(base);
    f.q[1] = *(const u32x4*)(const void*)(base + 16);
}
__device__ __forceinline__ void cvt_frag_f32(Frag& f, const float* base) {
    f32x4 c0 = *(const f32x4*)(const void*)(base);
    f32x4 c1 = *(const f32x4*)(const void*)(base + 4);
    f32x4 c2 = *(const f32x4*)(const void*)(base + 16);
    f32x4 c3 = *(const f32x4*)(const void*)(base + 20);
    f.h[0]=(_Float16)c0.x; f.h[1]=(_Float16)c0.y; f.h[2]=(_Float16)c0.z; f.h[3]=(_Float16)c0.w;
    f.h[4]=(_Float16)c1.x; f.h[5]=(_Float16)c1.y; f.h[6]=(_Float16)c1.z; f.h[7]=(_Float16)c1.w;
    f.h[8]=(_Float16)c2.x; f.h[9]=(_Float16)c2.y; f.h[10]=(_Float16)c2.z; f.h[11]=(_Float16)c2.w;
    f.h[12]=(_Float16)c3.x; f.h[13]=(_Float16)c3.y; f.h[14]=(_Float16)c3.z; f.h[15]=(_Float16)c3.w;
}

// LDS byte offset of a generic pointer into the workgroup's LDS allocation.
__device__ __forceinline__ unsigned lds_off(const void* p) {
    return (unsigned)(unsigned long long)
        (__attribute__((address_space(3))) const char*)p;
}

template <int N>
__device__ __forceinline__ void wait_tensor() {
#if HAVE_WTC
    __builtin_amdgcn_s_wait_tensorcnt(N);     // N is an ICE (template parameter)
#else
    if (N == 0) asm volatile("s_wait_tensorcnt 0x0" ::: "memory");
    else        asm volatile("s_wait_tensorcnt 0x2" ::: "memory");
#endif
}

#if HAVE_TDM
// Issue one TDM 2D tile DMA: global (row-major, row_stride elems) -> packed LDS tile.
// D# bit layout per CDNA5 ISA ch.8: group0 {count, lds_addr, global_addr, type=2},
// group1 {data_size, tensor dims, tile dims, dim0 stride}.
__device__ __forceinline__ void tdm_load_tile(unsigned lds_byte, const void* gsrc,
                                              int data_size_code,   // 0:1B 1:2B 2:4B
                                              unsigned tile_d0, unsigned tile_d1,
                                              unsigned long long tensor_d0,
                                              unsigned long long tensor_d1,
                                              unsigned long long row_stride) {
    unsigned long long ga = (unsigned long long)gsrc;
    v4u g0 = {0u, 0u, 0u, 0u};
    g0[0] = 1u;                                            // count=1 (valid)
    g0[1] = lds_byte;                                      // lds_addr
    g0[2] = (unsigned)(ga & 0xFFFFFFFFull);                // global_addr[31:0]
    g0[3] = (unsigned)((ga >> 32) & 0x01FFFFFFull) | (2u << 30);   // addr[56:32] | type=2
    v8i g1 = {0, 0, 0, 0, 0, 0, 0, 0};
    g1[0] = data_size_code << 16;                          // wg_mask=0 | data_size
    g1[1] = (int)((tensor_d0 & 0xFFFFull) << 16);          // tensor_dim0[15:0]
    g1[2] = (int)(((tensor_d0 >> 16) & 0xFFFFull) |
                  ((tensor_d1 & 0xFFFFull) << 16));        // dim0[31:16] | dim1[15:0]
    g1[3] = (int)(((tensor_d1 >> 16) & 0xFFFFull) |
                  ((unsigned long long)tile_d0 << 16));    // dim1[31:16] | tile_dim0
    g1[4] = (int)(tile_d1 & 0xFFFFu);                      // tile_dim1 | tile_dim2=0
    g1[5] = (int)(row_stride & 0xFFFFFFFFull);             // dim0_stride[31:0]
    g1[6] = (int)((row_stride >> 32) & 0xFFFFull);         // dim0_stride[47:32]
    g1[7] = 0;
    v4i gz = {0, 0, 0, 0};
#if defined(__clang_major__) && __clang_major__ >= 23
    v8i gz8 = {0, 0, 0, 0, 0, 0, 0, 0};
    __builtin_amdgcn_tensor_load_to_lds(g0, g1, gz, gz, gz8, 0);
#else
    __builtin_amdgcn_tensor_load_to_lds(g0, g1, gz, gz, 0);
#endif
}
#endif

// ---------------- Kernel 1: center + f16 convert ----------------
__global__ __launch_bounds__(256)
void prep_xc(const float* __restrict__ x, const float* __restrict__ pre_bias,
             _Float16* __restrict__ xc) {
    int i = blockIdx.x * 256 + threadIdx.x;
    xc[i] = (_Float16)(x[i] - pre_bias[i & (DMODEL - 1)]);
}

// ---------------- GEMM  C[M,N] = A[M,K] * B[N,K]^T + bias[N] ----------------
// A f16 (lda=K), B f32 streamed+converted, C f32 (ldc). Tile 128x128, BK=32,
// TDM double-buffered LDS staging, 8 waves x (4 M-frags x 2 N-frags).
__global__ __launch_bounds__(256)
void gemm_f16(const _Float16* __restrict__ A, const float* __restrict__ B,
              float* __restrict__ C, const float* __restrict__ bias,
              int K, int ldc, int arows, int brows) {
    __shared__ __align__(16) _Float16 sA[2][128 * 32];   // 2 x 8 KB
    __shared__ __align__(16) float    sB[2][128 * 32];   // 2 x 16 KB

    const int lane = threadIdx.x & 31;
    const int wave = threadIdx.x >> 5;
    const int mg   = wave & 1;
    const int ng   = wave >> 1;
    const int m0   = blockIdx.y * 128;
    const int n0   = blockIdx.x * 128;
    const int rin  = lane & 15;
    const int ksel = (lane >> 4) << 3;
    const int NK   = K / 32;

    v8f acc[4][2];
    const v8f vzero = {0.f,0.f,0.f,0.f,0.f,0.f,0.f,0.f};
#pragma unroll
    for (int i = 0; i < 4; ++i)
#pragma unroll
        for (int j = 0; j < 2; ++j) acc[i][j] = vzero;

    auto issue = [&](int buf, int k0) {
#if HAVE_TDM
        if (wave == 0) {
            tdm_load_tile(lds_off(&sA[buf][0]), A + (size_t)m0 * K + k0, 1, 32, 128,
                          (unsigned long long)(K - k0), (unsigned long long)(arows - m0),
                          (unsigned long long)K);
            tdm_load_tile(lds_off(&sB[buf][0]), B + (size_t)n0 * K + k0, 2, 32, 128,
                          (unsigned long long)(K - k0), (unsigned long long)(brows - n0),
                          (unsigned long long)K);
        }
#else
        // Cooperative fallback: 256 threads stage both tiles.
        const int t = threadIdx.x;
#pragma unroll
        for (int c = 0; c < 2; ++c) {                 // A: 512 x 16B
            int idx = t + c * 256;
            int row = idx >> 2, col = (idx & 3) * 8;
            *(u32x4*)&sA[buf][row * 32 + col] =
                *(const u32x4*)(const void*)(A + (size_t)(m0 + row) * K + k0 + col);
        }
#pragma unroll
        for (int c = 0; c < 4; ++c) {                 // B: 1024 x 16B
            int idx = t + c * 256;
            int row = idx >> 3, col = (idx & 7) * 4;
            *(f32x4*)&sB[buf][row * 32 + col] =
                *(const f32x4*)(const void*)(B + (size_t)(n0 + row) * K + k0 + col);
        }
#endif
    };

    issue(0, 0);
    for (int kt = 0; kt < NK; ++kt) {
        const int cur = kt & 1;
        if (kt + 1 < NK) {
            issue(cur ^ 1, (kt + 1) * 32);
#if HAVE_TDM
            if (wave == 0) wait_tensor<2>();   // current tiles done, next 2 in flight
#endif
        } else {
#if HAVE_TDM
            if (wave == 0) wait_tensor<0>();
#endif
        }
        __syncthreads();

        Frag a[4], b[2];
#pragma unroll
        for (int i = 0; i < 4; ++i) {
            int row = (mg * 4 + i) * 16 + rin;
            load_frag_f16(a[i], &sA[cur][row * 32 + ksel]);
        }
#pragma unroll
        for (int j = 0; j < 2; ++j) {
            int row = (ng * 2 + j) * 16 + rin;
            cvt_frag_f32(b[j], &sB[cur][row * 32 + ksel]);
        }
#pragma unroll
        for (int i = 0; i < 4; ++i)
#pragma unroll
            for (int j = 0; j < 2; ++j)
                acc[i][j] = __builtin_amdgcn_wmma_f32_16x16x32_f16(
                    false, a[i].v, false, b[j].v, (short)0, acc[i][j], false, false);
        __syncthreads();
    }

    const int moff = (lane >> 4) << 3;
#pragma unroll
    for (int i = 0; i < 4; ++i) {
#pragma unroll
        for (int j = 0; j < 2; ++j) {
            int mb = m0 + (mg * 4 + i) * 16 + moff;
            int nc = n0 + (ng * 2 + j) * 16 + rin;
            float bv = bias ? bias[nc] : 0.f;
#pragma unroll
            for (int r = 0; r < 8; ++r)
                C[(size_t)(mb + r) * ldc + nc] = acc[i][j][r] + bv;
        }
    }
}

// ---------------- wave32 argmax reduction ----------------
__device__ __forceinline__ void wave_argmax(float& v, int& i) {
#pragma unroll
    for (int d = 16; d > 0; d >>= 1) {
        float ov = __shfl_xor(v, d);
        int   oi = __shfl_xor(i, d);
        if (ov > v) { v = ov; i = oi; }
    }
}

// Generic iterative top-SEL extraction over an LDS-resident row.
template <int SEL>
__device__ __forceinline__ void select_topk(float* sv, float* rv, int* ri, int t,
                                            float* out_vals, int* out_inds) {
    const int lane = t & 31, wave = t >> 5;
    for (int sel = 0; sel < SEL; ++sel) {
        float best = -1e30f; int bi = 0;
        for (int i = t * 4; i < NDIRS; i += 1024) {          // vector LDS scan
            f32x4 v = *(const f32x4*)(const void*)&sv[i];
            if (v.x > best) { best = v.x; bi = i; }
            if (v.y > best) { best = v.y; bi = i + 1; }
            if (v.z > best) { best = v.z; bi = i + 2; }
            if (v.w > best) { best = v.w; bi = i + 3; }
        }
        wave_argmax(best, bi);
        if (lane == 0) { rv[wave] = best; ri[wave] = bi; }
        __syncthreads();
        if (t == 0) {
            float bv = rv[0]; int bidx = ri[0];
#pragma unroll
            for (int w = 1; w < 8; ++w)
                if (rv[w] > bv) { bv = rv[w]; bidx = ri[w]; }
            sv[bidx] = -1e30f;
            out_vals[sel] = bv;
            out_inds[sel] = bidx;
        }
        __syncthreads();
    }
}

// ---------------- Kernel 3: top-32 per row -> fired + f16 scatter ----------
__global__ __launch_bounds__(256)
void topk32_kernel(const float* __restrict__ latents, int* __restrict__ fired,
                   _Float16* __restrict__ dense) {
    extern __shared__ char smem[];
    float* sv = (float*)smem;
    float* rv = sv + NDIRS;
    int*   ri = (int*)(rv + 8);
    float* selv = (float*)(ri + 8);     // 32 selected vals
    int*   seli = (int*)(selv + TOPK);  // 32 selected inds
    const int b = blockIdx.x, t = threadIdx.x;
    const float* row = latents + (size_t)b * NDIRS;

    for (int i = t; i < NDIRS; i += 256) sv[i] = row[i];
    __syncthreads();

    select_topk<TOPK>(sv, rv, ri, t, selv, seli);

    if (t < TOPK) {
        float v = selv[t]; int idx = seli[t];
        if (v > EPSV) fired[idx] = 1;                         // benign races: all write 1
        float rl = v > 0.f ? v : 0.f;
        dense[(size_t)b * NDIRS + idx] = (_Float16)rl;
    }
}

// ---------------- Kernel 4: stats update + dead mask ----------------
__global__ __launch_bounds__(256)
void stats_kernel(const int* __restrict__ stats_in, const int* __restrict__ fired,
                  int* __restrict__ stats_out, int* __restrict__ dead) {
    int n = blockIdx.x * 256 + threadIdx.x;
    int ns = fired[n] ? 1 : stats_in[n] + 1;
    stats_out[n] = ns;
    dead[n] = (ns > DEADTH) ? 1 : 0;
}

// ---------------- Kernel 5: masked top-256 per row (auxk) ----------------
__global__ __launch_bounds__(256)
void auxk_kernel(const float* __restrict__ latents, const int* __restrict__ dead,
                 float* __restrict__ auxv, int* __restrict__ auxi) {
    extern __shared__ char smem[];
    float* sv = (float*)smem;
    float* rv = sv + NDIRS;
    int*   ri = (int*)(rv + 8);
    float* selv = (float*)(ri + 8);
    int*   seli = (int*)(selv + AUXK);
    const int b = blockIdx.x, t = threadIdx.x;
    const float* row = latents + (size_t)b * NDIRS;

    for (int i = t; i < NDIRS; i += 256) sv[i] = dead[i] ? row[i] : 0.0f;
    __syncthreads();

    select_topk<AUXK>(sv, rv, ri, t, selv, seli);

    for (int s = t; s < AUXK; s += 256) {
        float v = selv[s];
        auxv[(size_t)b * AUXK + s] = v > 0.f ? v : 0.f;       // relu
        auxi[(size_t)b * AUXK + s] = seli[s];
    }
}

// ---------------- Host-side orchestration ----------------
extern "C" void kernel_launch(void* const* d_in, const int* in_sizes, int n_in,
                              void* d_out, int out_size, void* d_ws, size_t ws_size,
                              hipStream_t stream) {
    const float* x           = (const float*)d_in[0];
    const float* enc_w       = (const float*)d_in[1];
    const float* dec_w       = (const float*)d_in[2];
    const float* pre_bias    = (const float*)d_in[3];
    const float* latent_bias = (const float*)d_in[4];
    const int*   stats_in    = (const int*)d_in[5];

    float* out_recons = (float*)d_out;
    float* out_auxv   = out_recons + (size_t)BATCH * DMODEL;
    int*   out_auxi   = (int*)(out_auxv + (size_t)BATCH * AUXK);
    int*   out_stats  = out_auxi + (size_t)BATCH * AUXK;

    char* w = (char*)d_ws;
    const size_t XC_B    = (size_t)BATCH * DMODEL * 2;
    const size_t LAT_B   = (size_t)BATCH * NDIRS * 4;
    const size_t DENSE_B = (size_t)BATCH * NDIRS * 2;
    _Float16* xc    = (_Float16*)(w);
    float*    lat   = (float*)(w + XC_B);
    _Float16* dense = (_Float16*)(w + XC_B + LAT_B);
    int*      fired = (int*)(w + XC_B + LAT_B + DENSE_B);
    int*      dead  = fired + NDIRS;

    (void)hipMemsetAsync(dense, 0, DENSE_B, stream);
    (void)hipMemsetAsync(fired, 0, (size_t)NDIRS * 4, stream);

    prep_xc<<<(BATCH * DMODEL) / 256, 256, 0, stream>>>(x, pre_bias, xc);

    // encode: latents = xc @ enc_w^T + latent_bias  (M=1024,N=32768,K=2048)
    gemm_f16<<<dim3(NDIRS / 128, BATCH / 128), 256, 0, stream>>>(
        xc, enc_w, lat, latent_bias, DMODEL, NDIRS, BATCH, NDIRS);

    const size_t SMEM = (size_t)NDIRS * 4 + 16 * 4 + (size_t)(AUXK) * 8 + 64;
    topk32_kernel<<<BATCH, 256, SMEM, stream>>>(lat, fired, dense);

    stats_kernel<<<NDIRS / 256, 256, 0, stream>>>(stats_in, fired, out_stats, dead);

    auxk_kernel<<<BATCH, 256, SMEM, stream>>>(lat, dead, out_auxv, out_auxi);

    // decode: recons = dense @ dec_w^T + pre_bias  (M=1024,N=2048,K=32768)
    gemm_f16<<<dim3(DMODEL / 128, BATCH / 128), 256, 0, stream>>>(
        dense, dec_w, out_recons, pre_bias, NDIRS, DMODEL, BATCH, DMODEL);
}
